// endCLS_normal_QK_model_88811333747464
// MI455X (gfx1250) — compile-verified, matrix operand order checked by reading, not run on hardware
//
#include <hip/hip_runtime.h>
#include <hip/hip_bf16.h>

#define G_    512
#define NMAX_ 32
#define EPER_ 64
#define D_    72
#define DE_   36
#define H_    6
#define L_    5
#define DS_   128
#define N_    (G_*NMAX_)
#define ET_   (G_*EPER_)
#define B_    (G_/2)
#define DH_   (D_/H_)
#define DF_   (8*D_)      /* 576 */
#define EPSF  1e-6f

#define ACT_NONE 0
#define ACT_SIG  1
#define ACT_RELU 2

typedef __attribute__((ext_vector_type(16))) __bf16    v16bf;
typedef __attribute__((ext_vector_type(8)))  float     v8f;
typedef __attribute__((ext_vector_type(8)))  unsigned  v8u;

__device__ __forceinline__ __bf16 f2bf(float f) { return (__bf16)f; }

// Byte-pair start (in halfwords) of fragment VGPR v for a 16-bit WMMA operand.
// ISA 7.12.2: lanes 0-15 hold K=0..7 (VGPR0-3) + K=16..23 (VGPR4-7);
// lanes 16-31 hold K=8..15 + K=24..31.  Pairs (2v,2v+1) are K-consecutive.
__device__ __forceinline__ int kofs(int v, int half) {
  return half * 8 + ((v < 4) ? (v * 2) : (16 + (v - 4) * 2));
}

// Build a v16bf fragment from a bf16 row whose K-dim is contiguous.
__device__ __forceinline__ v16bf frag_from_row(const __bf16* row, int half) {
  v8u u;
  #pragma unroll
  for (int v = 0; v < 8; v++)
    u[v] = *(const unsigned*)(row + kofs(v, half));   // 4B-aligned (kofs even)
  return __builtin_bit_cast(v16bf, u);
}

// ---------------------------------------------------------------------------
// Generic tiled WMMA GEMM: C[M,N] = epi(A[M,K] @ B[K,N] + bias) * rowscale
// fp32 in memory -> bf16 LDS tiles (converted once) -> bf16 WMMA, f32 acc.
// Block: 128 threads = 4 waves; tile 64(M) x 80(N); K chunks of 32.
// ---------------------------------------------------------------------------
__global__ __launch_bounds__(128) void wmma_gemm(
    const float* __restrict__ A, const float* __restrict__ Bm,
    const float* __restrict__ bias, const float* __restrict__ rowscale,
    float* __restrict__ C, int M, int K, int Nn, int act)
{
  __shared__ __align__(16) __bf16 As[64][34];   // [m][k]
  __shared__ __align__(16) __bf16 Bt[80][34];   // [n][k]  (transposed)
  const int row0 = blockIdx.x * 64;
  const int col0 = blockIdx.y * 80;
  const int tid  = threadIdx.x;
  const int wave = tid >> 5, lane = tid & 31;
  const int nr = lane & 15, half = lane >> 4;

  v8f acc[5];
  #pragma unroll
  for (int t = 0; t < 5; t++)
    #pragma unroll
    for (int r = 0; r < 8; r++) acc[t][r] = 0.f;

  const int KC = (K + 31) >> 5;
  for (int kc = 0; kc < KC; kc++) {
    const int k0 = kc << 5;
    if (kc + 1 < KC) {                       // prefetch next A chunk
      int r = row0 + (tid & 63);
      if (r < M) __builtin_prefetch(&A[(size_t)r * K + k0 + 32], 0, 0);
    }
    for (int idx = tid; idx < 64 * 32; idx += 128) {
      int m = idx >> 5, j = idx & 31;
      int r = row0 + m, k = k0 + j;
      As[m][j] = f2bf((r < M && k < K) ? A[(size_t)r * K + k] : 0.f);
    }
    for (int idx = tid; idx < 32 * 80; idx += 128) {
      int kk = idx / 80, n = idx % 80;     // n fastest -> coalesced global reads
      int k = k0 + kk, c = col0 + n;
      Bt[n][kk] = f2bf((k < K && c < Nn) ? Bm[(size_t)k * Nn + c] : 0.f);
    }
    __syncthreads();

    v16bf af = frag_from_row(&As[wave * 16 + nr][0], half);
    #pragma unroll
    for (int t = 0; t < 5; t++) {
      v16bf bfg = frag_from_row(&Bt[t * 16 + nr][0], half);
      acc[t] = __builtin_amdgcn_wmma_f32_16x16x32_bf16(
          false, af, false, bfg, (short)0, acc[t], false, false);
    }
    __syncthreads();
  }

  #pragma unroll
  for (int t = 0; t < 5; t++) {
    int col = col0 + t * 16 + nr;
    float bval = (bias != nullptr && col < Nn) ? bias[col] : 0.f;
    #pragma unroll
    for (int r = 0; r < 8; r++) {
      int row = row0 + wave * 16 + half * 8 + r;
      if (row < M && col < Nn) {
        float v = acc[t][r] + bval;
        if (act == ACT_SIG)  v = 1.f / (1.f + __expf(-v));
        if (act == ACT_RELU) v = fmaxf(v, 0.f);
        if (rowscale) v *= rowscale[row];
        C[(size_t)row * Nn + col] = v;
      }
    }
  }
}

// ---------------------------------------------------------------------------
// ECC as a Khatri-Rao WMMA GEMM:
// msg[e,o] = edge_mask[e] * sum_{f,i} Ee[e,f] * Xn[senders[e],i] * K[f,i,o]
// Inner dim padded: i -> 96 (3 chunks of 32), K' = 36*96 = 3456.
// A-tile built on the fly in LDS (bf16); B-tile from reshaped ecc kernel.
// ---------------------------------------------------------------------------
__global__ __launch_bounds__(128) void wmma_gemm_ecc(
    const float* __restrict__ Ee, const float* __restrict__ Xn,
    const int* __restrict__ senders, const float* __restrict__ ecc,
    const float* __restrict__ emask, float* __restrict__ msg, int M)
{
  __shared__ __align__(16) __bf16 As[64][34];
  __shared__ __align__(16) __bf16 Bt[80][34];
  const int row0 = blockIdx.x * 64;
  const int tid  = threadIdx.x;
  const int wave = tid >> 5, lane = tid & 31;
  const int nr = lane & 15, half = lane >> 4;

  v8f acc[5];
  #pragma unroll
  for (int t = 0; t < 5; t++)
    #pragma unroll
    for (int r = 0; r < 8; r++) acc[t][r] = 0.f;

  for (int kc = 0; kc < 108; kc++) {        // 36 feats * 3 chunks of 32 (i pad 96)
    const int f  = kc / 3;
    const int i0 = (kc % 3) * 32;
    for (int idx = tid; idx < 64 * 32; idx += 128) {
      int m = idx >> 5, j = idx & 31;
      int row = row0 + m, i = i0 + j;
      float v = 0.f;
      if (row < M && i < D_)
        v = Ee[(size_t)row * DE_ + f] * Xn[(size_t)senders[row] * D_ + i];
      As[m][j] = f2bf(v);
    }
    for (int idx = tid; idx < 32 * 80; idx += 128) {
      int kk = idx / 80, n = idx % 80;
      int i = i0 + kk;
      Bt[n][kk] = f2bf((i < D_ && n < D_) ? ecc[((size_t)f * D_ + i) * D_ + n] : 0.f);
    }
    __syncthreads();

    v16bf af = frag_from_row(&As[wave * 16 + nr][0], half);
    #pragma unroll
    for (int t = 0; t < 5; t++) {
      v16bf bfg = frag_from_row(&Bt[t * 16 + nr][0], half);
      acc[t] = __builtin_amdgcn_wmma_f32_16x16x32_bf16(
          false, af, false, bfg, (short)0, acc[t], false, false);
    }
    __syncthreads();
  }

  #pragma unroll
  for (int t = 0; t < 5; t++) {
    int col = t * 16 + nr;
    #pragma unroll
    for (int r = 0; r < 8; r++) {
      int row = row0 + wave * 16 + half * 8 + r;
      if (row < M && col < D_)
        msg[(size_t)row * D_ + col] = acc[t][r] * emask[row];
    }
  }
}

// ---------------------------------------------------------------------------
// Node / edge embeddings
// ---------------------------------------------------------------------------
__global__ void node_embed(const float* __restrict__ X, const float* __restrict__ ea,
                           const float* __restrict__ ec, const float* __restrict__ eh,
                           const float* __restrict__ Wx, const float* __restrict__ bx,
                           const float* __restrict__ nm, float* __restrict__ Xn)
{
  int n = blockIdx.x, d = threadIdx.x;
  if (d >= D_) return;
  int ai = (int)X[n * 8 + 0], ci = (int)X[n * 8 + 1], hi = (int)X[n * 8 + 2];
  float acc = ea[ai * D_ + d] + ec[ci * D_ + d] + eh[hi * D_ + d] + bx[d];
  #pragma unroll
  for (int j = 0; j < 5; j++) acc += X[n * 8 + 3 + j] * Wx[j * D_ + d];
  Xn[(size_t)n * D_ + d] = acc * nm[n];
}

__global__ void edge_embed(const float* __restrict__ E, const float* __restrict__ eb,
                           const float* __restrict__ We, const float* __restrict__ be,
                           const float* __restrict__ em, float* __restrict__ Ee)
{
  int e = blockIdx.x, d = threadIdx.x;
  if (d >= DE_) return;
  int bi = (int)E[e * 2 + 0];
  float v = eb[bi * DE_ + d] + E[e * 2 + 1] * We[d] + be[d];
  Ee[(size_t)e * DE_ + d] = v * em[e];
}

// ---------------------------------------------------------------------------
// 6-step MPNN, one graph per workgroup, all in LDS (edges are intra-graph).
// ---------------------------------------------------------------------------
__global__ __launch_bounds__(256) void mpnn_kernel(
    const float* __restrict__ Xin, const float* __restrict__ gate,
    const int* __restrict__ senders, const int* __restrict__ receivers,
    const float* __restrict__ nmask, float* __restrict__ Xout)
{
  int g = blockIdx.x, tid = threadIdx.x;
  __shared__ float h[NMAX_][D_ + 1];
  __shared__ float agg[NMAX_][D_ + 1];
  __shared__ int   sl[EPER_], rl[EPER_];
  __shared__ float nm[NMAX_];

  for (int idx = tid; idx < NMAX_ * D_; idx += 256) {
    int n = idx / D_, d = idx % D_;
    h[n][d] = Xin[((size_t)g * NMAX_ + n) * D_ + d];
  }
  if (tid < EPER_) {
    sl[tid] = senders[g * EPER_ + tid] - g * NMAX_;
    rl[tid] = receivers[g * EPER_ + tid] - g * NMAX_;
  }
  if (tid < NMAX_) nm[tid] = nmask[g * NMAX_ + tid];
  __syncthreads();

  for (int step = 0; step < 6; step++) {
    for (int idx = tid; idx < NMAX_ * D_; idx += 256) agg[idx / D_][idx % D_] = 0.f;
    __syncthreads();
    for (int idx = tid; idx < EPER_ * D_; idx += 256) {
      int e = idx / D_, d = idx % D_;
      float m = h[sl[e]][d] * gate[((size_t)g * EPER_ + e) * D_ + d];
      atomicAdd(&agg[rl[e]][d], m);
    }
    __syncthreads();
    for (int idx = tid; idx < NMAX_ * D_; idx += 256) {
      int n = idx / D_, d = idx % D_;
      h[n][d] = (h[n][d] + agg[n][d]) * nm[n];
    }
    __syncthreads();
  }
  for (int idx = tid; idx < NMAX_ * D_; idx += 256) {
    int n = idx / D_, d = idx % D_;
    Xout[((size_t)g * NMAX_ + n) * D_ + d] = h[n][d];
  }
}

// ---------------------------------------------------------------------------
// Masked multi-head attention, one graph per workgroup (32x32, 6 heads, DH=12)
// ---------------------------------------------------------------------------
__global__ __launch_bounds__(256) void attn_kernel(
    const float* __restrict__ q, const float* __restrict__ k,
    const float* __restrict__ v, const float* __restrict__ nmask,
    float* __restrict__ out)
{
  int g = blockIdx.x, tid = threadIdx.x;
  __shared__ float qs[NMAX_][D_], ks[NMAX_][D_], vs[NMAX_][D_];
  __shared__ float km[NMAX_];
  for (int idx = tid; idx < NMAX_ * D_; idx += 256) {
    int n = idx / D_, d = idx % D_;
    size_t off = ((size_t)g * NMAX_ + n) * D_ + d;
    qs[n][d] = q[off]; ks[n][d] = k[off]; vs[n][d] = v[off];
  }
  if (tid < NMAX_) km[tid] = nmask[g * NMAX_ + tid];
  __syncthreads();

  if (tid < NMAX_ * H_) {
    int i = tid % NMAX_, hh = tid / NMAX_;
    int d0 = hh * DH_;
    const float scale = rsqrtf((float)DH_);
    float sc[NMAX_], mx = -1e30f;
    for (int j = 0; j < NMAX_; j++) {
      float s = 0.f;
      #pragma unroll
      for (int d = 0; d < DH_; d++) s += qs[i][d0 + d] * ks[j][d0 + d];
      s = (km[j] > 0.f) ? s * scale : -1e9f;
      sc[j] = s; mx = fmaxf(mx, s);
    }
    float sum = 0.f;
    for (int j = 0; j < NMAX_; j++) { sc[j] = __expf(sc[j] - mx); sum += sc[j]; }
    float inv = 1.f / sum;
    #pragma unroll
    for (int d = 0; d < DH_; d++) {
      float o = 0.f;
      for (int j = 0; j < NMAX_; j++) o += sc[j] * vs[j][d0 + d];
      out[((size_t)g * NMAX_ + i) * D_ + d0 + d] = o * inv;
    }
  }
}

// ---------------------------------------------------------------------------
// LayerNorm over D=72 with optional residual and node-mask (1 thread / row)
// ---------------------------------------------------------------------------
__global__ void ln_kernel(const float* __restrict__ x, const float* __restrict__ res,
                          const float* __restrict__ s, const float* __restrict__ b,
                          const float* __restrict__ nm, float* __restrict__ out, int M)
{
  int row = blockIdx.x * blockDim.x + threadIdx.x;
  if (row >= M) return;
  float sum = 0.f, sq = 0.f;
  for (int d = 0; d < D_; d++) {
    float v = x[(size_t)row * D_ + d] + (res ? res[(size_t)row * D_ + d] : 0.f);
    sum += v; sq += v * v;
  }
  float mean = sum * (1.f / D_);
  float var  = sq * (1.f / D_) - mean * mean;
  float inv  = rsqrtf(var + EPSF);
  float m    = nm ? nm[row] : 1.f;
  for (int d = 0; d < D_; d++) {
    float v = x[(size_t)row * D_ + d] + (res ? res[(size_t)row * D_ + d] : 0.f);
    out[(size_t)row * D_ + d] = ((v - mean) * inv * s[d] + b[d]) * m;
  }
}

// ---------------------------------------------------------------------------
// ECC combine: Xn = (Xroot + segment_sum(msg by receiver)) * nm   (per graph)
// ---------------------------------------------------------------------------
__global__ __launch_bounds__(256) void ecc_combine(
    const float* __restrict__ Xroot, const float* __restrict__ msg,
    const int* __restrict__ receivers, const float* __restrict__ nmask,
    float* __restrict__ Xout)
{
  int g = blockIdx.x, tid = threadIdx.x;
  __shared__ float agg[NMAX_][D_ + 1];
  for (int idx = tid; idx < NMAX_ * D_; idx += 256) agg[idx / D_][idx % D_] = 0.f;
  __syncthreads();
  for (int idx = tid; idx < EPER_ * D_; idx += 256) {
    int e = idx / D_, d = idx % D_;
    int r = receivers[g * EPER_ + e] - g * NMAX_;
    atomicAdd(&agg[r][d], msg[((size_t)g * EPER_ + e) * D_ + d]);
  }
  __syncthreads();
  for (int idx = tid; idx < NMAX_ * D_; idx += 256) {
    int n = idx / D_, d = idx % D_;
    size_t off = ((size_t)g * NMAX_ + n) * D_ + d;
    Xout[off] = (Xroot[off] + agg[n][d]) * nmask[g * NMAX_ + n];
  }
}

// ---------------------------------------------------------------------------
// Masked attention pooling per graph
// ---------------------------------------------------------------------------
__global__ __launch_bounds__(128) void pool_kernel(
    const float* __restrict__ Xn, const float* __restrict__ poolW,
    const float* __restrict__ nmask, float* __restrict__ pooled)
{
  int g = blockIdx.x, tid = threadIdx.x;
  __shared__ float a[NMAX_];
  if (tid < NMAX_) {
    int node = g * NMAX_ + tid;
    float s = 0.f;
    for (int d = 0; d < D_; d++) s += Xn[(size_t)node * D_ + d] * poolW[d];
    a[tid] = (nmask[node] > 0.f) ? s : -1e9f;
  }
  __syncthreads();
  if (tid == 0) {
    float mx = -1e30f;
    for (int j = 0; j < NMAX_; j++) mx = fmaxf(mx, a[j]);
    float sum = 0.f;
    for (int j = 0; j < NMAX_; j++) { a[j] = __expf(a[j] - mx); sum += a[j]; }
    float inv = 1.f / sum;
    for (int j = 0; j < NMAX_; j++) a[j] *= inv;
  }
  __syncthreads();
  for (int d = tid; d < D_; d += 128) {
    float s = 0.f;
    for (int j = 0; j < NMAX_; j++) s += a[j] * Xn[((size_t)g * NMAX_ + j) * D_ + d];
    pooled[(size_t)g * D_ + d] = s;
  }
}

__global__ void concat_kernel(const float* __restrict__ pooled,
                              const float* __restrict__ S, float* __restrict__ xcat)
{
  int b = blockIdx.x, t = threadIdx.x;
  if (t < D_)                xcat[(size_t)b * (D_ + DS_) + t] = pooled[(size_t)(2 * b) * D_ + t];
  else if (t < D_ + DS_)     xcat[(size_t)b * (D_ + DS_) + t] = S[(size_t)b * DS_ + (t - D_)];
}

__global__ void final_out(const float* __restrict__ h2, const float* __restrict__ outW,
                          const float* __restrict__ outb, float* __restrict__ out)
{
  int b = blockIdx.x * blockDim.x + threadIdx.x;
  if (b >= B_) return;
  float s = outb[0];
  for (int d = 0; d < D_; d++) s += h2[(size_t)b * D_ + d] * outW[d];
  out[b] = s;
}

// ---------------------------------------------------------------------------
static inline void launch_gemm(const float* A, const float* B, const float* bias,
                               const float* rs, float* C, int M, int K, int N,
                               int act, hipStream_t s)
{
  dim3 grid((M + 63) / 64, (N + 79) / 80);
  wmma_gemm<<<grid, 128, 0, s>>>(A, B, bias, rs, C, M, K, N, act);
}

extern "C" void kernel_launch(void* const* d_in, const int* in_sizes, int n_in,
                              void* d_out, int out_size, void* d_ws, size_t ws_size,
                              hipStream_t stream)
{
  (void)in_sizes; (void)n_in; (void)out_size; (void)ws_size;
  const float* S          = (const float*)d_in[0];
  const float* X          = (const float*)d_in[1];
  const float* E          = (const float*)d_in[2];
  const int*   senders    = (const int*)d_in[3];
  const int*   receivers  = (const int*)d_in[4];
  const float* node_mask  = (const float*)d_in[5];
  const float* edge_mask  = (const float*)d_in[6];
  const float* emb_atomic = (const float*)d_in[7];
  const float* emb_chiral = (const float*)d_in[8];
  const float* emb_hyb    = (const float*)d_in[9];
  const float* emb_bond   = (const float*)d_in[10];
  const float* Wx   = (const float*)d_in[11];
  const float* bx   = (const float*)d_in[12];
  const float* We   = (const float*)d_in[13];
  const float* be   = (const float*)d_in[14];
  const float* mpQ_Wg = (const float*)d_in[15];
  const float* mpQ_bg = (const float*)d_in[16];
  const float* mpK_Wg = (const float*)d_in[17];
  const float* mpK_bg = (const float*)d_in[18];
  const float* Wq = (const float*)d_in[19];
  const float* bq = (const float*)d_in[20];
  const float* Wk = (const float*)d_in[21];
  const float* bk = (const float*)d_in[22];
  const float* Wv = (const float*)d_in[23];
  const float* bv = (const float*)d_in[24];
  const float* Wo = (const float*)d_in[25];
  const float* bo = (const float*)d_in[26];
  const float* ln1s = (const float*)d_in[27];
  const float* ln1b = (const float*)d_in[28];
  const float* W1 = (const float*)d_in[29];
  const float* b1 = (const float*)d_in[30];
  const float* W2 = (const float*)d_in[31];
  const float* b2 = (const float*)d_in[32];
  const float* ln2s = (const float*)d_in[33];
  const float* ln2b = (const float*)d_in[34];
  const float* ecc_kernel = (const float*)d_in[35];
  const float* rootW = (const float*)d_in[36];
  const float* rootb = (const float*)d_in[37];
  const float* poolW = (const float*)d_in[38];
  const float* d1W = (const float*)d_in[39];
  const float* d1b = (const float*)d_in[40];
  const float* d2W = (const float*)d_in[41];
  const float* d2b = (const float*)d_in[42];
  const float* lnSs = (const float*)d_in[43];
  const float* lnSb = (const float*)d_in[44];
  const float* outW = (const float*)d_in[45];
  const float* outb = (const float*)d_in[46];

  // -------- workspace carve-out (fp32), with aliasing of dead regions ------
  float* ws = (float*)d_ws;
  size_t o = 0;
  float* Xn   = ws + o; o += (size_t)N_ * D_;        // persistent node state
  float* Ee   = ws + o; o += (size_t)ET_ * DE_;      // persistent edge feats
  float* Qn   = ws + o; o += (size_t)N_ * D_;
  float* Kn   = ws + o; o += (size_t)N_ * D_;
  float* R    = ws + o; o += (size_t)N_ * DF_;       // big region, aliased below
  float* oprj = ws + o; o += (size_t)N_ * D_;        // also reused as Xroot
  float* ffn2 = ws + o; o += (size_t)N_ * D_;
  float* pooled = ws + o; o += (size_t)G_ * D_;
  float* xcat = ws + o; o += (size_t)B_ * (D_ + DS_);
  float* h1   = ws + o; o += (size_t)B_ * 256;
  float* h2   = ws + o; o += (size_t)B_ * D_;

  float* gQ   = R;                                   // ET*72
  float* gK   = R + (size_t)ET_ * D_;                // ET*72
  float* qb   = gK + (size_t)ET_ * D_;               // N*72
  float* kb   = qb + (size_t)N_ * D_;
  float* vb   = kb + (size_t)N_ * D_;
  float* attn = vb + (size_t)N_ * D_;
  float* ffn1 = R;                                   // N*576, reuses whole R
  float* msg  = R;                                   // ET*72, after layers
  float* Xroot = oprj;

  // -------- embeddings -----------------------------------------------------
  node_embed<<<N_, 96, 0, stream>>>(X, emb_atomic, emb_chiral, emb_hyb, Wx, bx, node_mask, Xn);
  edge_embed<<<ET_, 64, 0, stream>>>(E, emb_bond, We, be, edge_mask, Ee);

  // -------- transformer layers --------------------------------------------
  for (int l = 0; l < L_; l++) {
    launch_gemm(Ee, mpQ_Wg + (size_t)l * DE_ * D_, mpQ_bg + l * D_, edge_mask,
                gQ, ET_, DE_, D_, ACT_SIG, stream);
    launch_gemm(Ee, mpK_Wg + (size_t)l * DE_ * D_, mpK_bg + l * D_, edge_mask,
                gK, ET_, DE_, D_, ACT_SIG, stream);
    mpnn_kernel<<<G_, 256, 0, stream>>>(Xn, gQ, senders, receivers, node_mask, Qn);
    mpnn_kernel<<<G_, 256, 0, stream>>>(Xn, gK, senders, receivers, node_mask, Kn);

    launch_gemm(Qn, Wq + (size_t)l * D_ * D_, bq + l * D_, nullptr, qb, N_, D_, D_, ACT_NONE, stream);
    launch_gemm(Kn, Wk + (size_t)l * D_ * D_, bk + l * D_, nullptr, kb, N_, D_, D_, ACT_NONE, stream);
    launch_gemm(Kn, Wv + (size_t)l * D_ * D_, bv + l * D_, nullptr, vb, N_, D_, D_, ACT_NONE, stream);

    attn_kernel<<<G_, 256, 0, stream>>>(qb, kb, vb, node_mask, attn);
    launch_gemm(attn, Wo + (size_t)l * D_ * D_, bo + l * D_, nullptr, oprj, N_, D_, D_, ACT_NONE, stream);

    ln_kernel<<<(N_ + 255) / 256, 256, 0, stream>>>(oprj, Xn, ln1s + l * D_, ln1b + l * D_,
                                                    node_mask, Xn, N_);
    launch_gemm(Xn, W1 + (size_t)l * D_ * DF_, b1 + l * DF_, nullptr, ffn1, N_, D_, DF_, ACT_RELU, stream);
    launch_gemm(ffn1, W2 + (size_t)l * DF_ * D_, b2 + l * D_, nullptr, ffn2, N_, DF_, D_, ACT_NONE, stream);
    ln_kernel<<<(N_ + 255) / 256, 256, 0, stream>>>(ffn2, Xn, ln2s + l * D_, ln2b + l * D_,
                                                    node_mask, Xn, N_);
  }

  // -------- ECC + readout --------------------------------------------------
  wmma_gemm_ecc<<<dim3(ET_ / 64, 1), 128, 0, stream>>>(Ee, Xn, senders, ecc_kernel,
                                                       edge_mask, msg, ET_);
  launch_gemm(Xn, rootW, rootb, nullptr, Xroot, N_, D_, D_, ACT_NONE, stream);
  ecc_combine<<<G_, 256, 0, stream>>>(Xroot, msg, receivers, node_mask, Xn);

  pool_kernel<<<G_, 128, 0, stream>>>(Xn, poolW, node_mask, pooled);
  concat_kernel<<<B_, 256, 0, stream>>>(pooled, S, xcat);

  launch_gemm(xcat, d1W, d1b, nullptr, h1, B_, D_ + DS_, 256, ACT_RELU, stream);
  launch_gemm(h1, d2W, d2b, nullptr, h2, B_, 256, D_, ACT_NONE, stream);
  ln_kernel<<<1, 256, 0, stream>>>(h2, nullptr, lnSs, lnSb, nullptr, h2, B_);
  final_out<<<1, 256, 0, stream>>>(h2, outW, outb, (float*)d_out);
}
